// MultiHeadAttention_76441827934939
// MI455X (gfx1250) — matrix operand order repaired and checked by reading, not compiled, not run on hardware
//
#include <hip/hip_runtime.h>

// ---------------------------------------------------------------------------
// MI455X (gfx1250) multi-head attention forward.
// bf16 WMMA (v_wmma_f32_16x16x32_bf16) everywhere; flash-attention inner loop;
// K/V tiles staged into LDS via the Tensor Data Mover (tensor_load_to_lds,
// TENSORcnt) when the toolchain exposes the builtin, else cooperative
// global->LDS staging.
// ---------------------------------------------------------------------------

typedef __attribute__((ext_vector_type(16))) __bf16       v16bf;
typedef __attribute__((ext_vector_type(8)))  float        v8f;
typedef __attribute__((ext_vector_type(4)))  unsigned int v4u;
typedef __attribute__((ext_vector_type(4)))  float        v4f;
typedef __attribute__((ext_vector_type(4)))  int          v4i;
typedef __attribute__((ext_vector_type(8)))  int          v8i;

#if defined(__has_builtin)
#if __has_builtin(__builtin_amdgcn_tensor_load_to_lds) && \
    __has_builtin(__builtin_amdgcn_s_wait_tensorcnt)
#define HAVE_TDM 1
#endif
#endif
#ifndef HAVE_TDM
#define HAVE_TDM 0
#endif

union Frag16 {
  v16bf  v;
  v4u    u4[2];
  __bf16 e[16];
};

constexpr int B_ = 2, S_ = 2048, E_ = 1024, H_ = 8, D_ = 128;

static __device__ __forceinline__ v8f wmma_bf16(const Frag16& a, const Frag16& b, v8f c) {
  // (neg_a, A, neg_b, B, c_mod, C, reuse_a, reuse_b)
  return __builtin_amdgcn_wmma_f32_16x16x32_bf16(false, a.v, false, b.v, (short)0, c, false, false);
}

#if HAVE_TDM
// Issue one 2-D TDM load: tile (tile_d0 x tile_d1) of bf16 from a row-major
// tensor (inner dim tensor_d0, row stride stride0 elements) into LDS at
// lds_addr, packed row-major with row length tile_d0. D# layout per CDNA5 ISA
// ch.8: group0 = {count/type/addrs}, group1 = {dims/tiles/strides}.
static __device__ __forceinline__ void tdm_load_2d_bf16(
    unsigned lds_addr, const __bf16* gptr,
    unsigned tensor_d0, unsigned tensor_d1,
    unsigned tile_d0, unsigned tile_d1,
    unsigned long long stride0) {
  const unsigned long long ga = (unsigned long long)(uintptr_t)gptr;
  v4u g0;
  g0[0] = 1u;                                        // count=1 (valid user D#)
  g0[1] = lds_addr;                                  // lds_addr [63:32]
  g0[2] = (unsigned)(ga & 0xFFFFFFFFu);              // global_addr lo
  g0[3] = (unsigned)((ga >> 32) & 0x01FFFFFFu)       // global_addr hi (57-bit)
          | (2u << 30);                              // type=2 ("image")
  v8i g1;
  g1[0] = (int)(1u << 16);                           // data_size=1 -> 2 bytes
  g1[1] = (int)((tensor_d0 & 0xFFFFu) << 16);        // tensor_dim0[15:0]
  g1[2] = (int)(((tensor_d0 >> 16) & 0xFFFFu) | ((tensor_d1 & 0xFFFFu) << 16));
  g1[3] = (int)(((tensor_d1 >> 16) & 0xFFFFu) | ((tile_d0 & 0xFFFFu) << 16));
  g1[4] = (int)(tile_d1 & 0xFFFFu);                  // tile_dim1; tile_dim2=0
  g1[5] = (int)(unsigned)(stride0 & 0xFFFFFFFFu);    // tensor_dim0_stride lo
  g1[6] = (int)(unsigned)((stride0 >> 32) & 0xFFFFu);// stride hi; dim1_stride=0
  g1[7] = 0;
  const v4i z4 = {0, 0, 0, 0};
#if __clang_major__ >= 23
  const v8i z8 = {0, 0, 0, 0, 0, 0, 0, 0};
  __builtin_amdgcn_tensor_load_to_lds(g0, g1, z4, z4, z8, 0);
#else
  __builtin_amdgcn_tensor_load_to_lds(g0, g1, z4, z4, 0);
#endif
}
#endif  // HAVE_TDM

// ---------------------------------------------------------------- weights cvt
__global__ void convert_f32_bf16(const float* __restrict__ src,
                                 __bf16* __restrict__ dst, int n) {
  int i = blockIdx.x * blockDim.x + threadIdx.x;
  if (i < n) dst[i] = (__bf16)src[i];
}

// ------------------------------------------------------------ QKV projection
// C[m,n] = X[m,:] @ W[:,n] + bias[n], then scaled and written bf16 into a
// head-major layout. kt_layout==1 writes K transposed [B,H,D,S].
// Block: 128 threads = 4 waves, each wave does a 16x64 output tile.
__global__ __launch_bounds__(128)
void gemm_qkv(const float* __restrict__ X, const __bf16* __restrict__ W,
              const float* __restrict__ bias, __bf16* __restrict__ out,
              int kt_layout, float out_scale) {
  const int lane = threadIdx.x & 31;
  const int wid  = threadIdx.x >> 5;
  const int hl   = lane >> 4;     // wave half
  const int l16  = lane & 15;
  const int m_base = blockIdx.x * 64 + wid * 16;
  const int n_base = blockIdx.y * 64;

  v8f acc[4] = {};
  const float* Arow = X + (size_t)(m_base + l16) * E_;

  for (int k0 = 0; k0 < E_; k0 += 32) {
    // A fragment: f32 -> bf16 in-register. Row m = l16; K chunks 8*hl and 16+8*hl.
    Frag16 af;
    const v4f* a0 = (const v4f*)(Arow + k0 + 8 * hl);
    const v4f* a1 = (const v4f*)(Arow + k0 + 16 + 8 * hl);
    v4f x0 = a0[0], x1 = a0[1], y0 = a1[0], y1 = a1[1];
#pragma unroll
    for (int j = 0; j < 4; ++j) {
      af.e[j]      = (__bf16)x0[j];
      af.e[4 + j]  = (__bf16)x1[j];
      af.e[8 + j]  = (__bf16)y0[j];
      af.e[12 + j] = (__bf16)y1[j];
    }
    // B fragments: column n = l16; K rows 8*hl+s and 16+8*hl+s.
    const __bf16* Wk0 = W + (size_t)k0 * E_ + n_base + l16;
#pragma unroll
    for (int nt = 0; nt < 4; ++nt) {
      Frag16 bf;
      for (int s = 0; s < 8; ++s) {
        bf.e[s]     = Wk0[(size_t)(8 * hl + s) * E_ + nt * 16];
        bf.e[8 + s] = Wk0[(size_t)(16 + 8 * hl + s) * E_ + nt * 16];
      }
      acc[nt] = wmma_bf16(af, bf, acc[nt]);
    }
  }

#pragma unroll
  for (int nt = 0; nt < 4; ++nt) {
    const int col = n_base + nt * 16 + l16;
    const float bv = bias[col];
    const int hh = col >> 7, dd = col & (D_ - 1);
#pragma unroll
    for (int r = 0; r < 8; ++r) {
      const int row = m_base + r + 8 * hl;         // [0, B*S)
      const int bb = row >> 11, ss = row & (S_ - 1);
      const float vv = (acc[nt][r] + bv) * out_scale;
      const size_t idx = kt_layout
        ? ((((size_t)bb * H_ + hh) * D_ + dd) * S_ + ss)   // [B,H,D,S]
        : ((((size_t)bb * H_ + hh) * S_ + ss) * D_ + dd);  // [B,H,S,D]
      out[idx] = (__bf16)vv;
    }
  }
}

// ------------------------------------------------------- flash attention core
// Grid (S/128, H, B); 256 threads = 8 waves; wave owns 16 query rows.
// Q/V: [B,H,S,D] bf16, Kt: [B,H,D,S] bf16. Output attn [B*S, E] bf16.
// All waves run a uniform key-block loop (so barriers are legal); each wave
// computes only blocks inside its causal bound. K/V tiles are staged in LDS
// once per block (TDM when available) instead of 8x redundant global streams.
__global__ __launch_bounds__(256)
void attn_fwd(const __bf16* __restrict__ Q, const __bf16* __restrict__ Kt,
              const __bf16* __restrict__ V, __bf16* __restrict__ O) {
  const int lane = threadIdx.x & 31;
  const int wid  = threadIdx.x >> 5;
  const int hl   = lane >> 4;
  const int l16  = lane & 15;
  const int h = blockIdx.y, b = blockIdx.z;
  const int q_base = blockIdx.x * 128 + wid * 16;

  const __bf16* Qp = Q + ((size_t)b * H_ + h) * (size_t)S_ * D_;
  const __bf16* Kp = Kt + ((size_t)b * H_ + h) * (size_t)D_ * S_;
  const __bf16* Vp = V + ((size_t)b * H_ + h) * (size_t)S_ * D_;

  __shared__ __align__(16) __bf16 K_lds[128][64];     // [d][key]   16 KB
  __shared__ __align__(16) __bf16 V_lds[64][128];     // [key][d]   16 KB
  __shared__ __align__(16) __bf16 P_lds[8][16][64];   // per-wave P 16 KB

  // Q fragments for this wave's 16 rows (already scaled by 1/sqrt(D) upstream).
  Frag16 qf[4];
  {
    const __bf16* qrow = Qp + (size_t)(q_base + l16) * D_;
#pragma unroll
    for (int kb = 0; kb < 4; ++kb) {
      qf[kb].u4[0] = *(const v4u*)(qrow + kb * 32 + 8 * hl);
      qf[kb].u4[1] = *(const v4u*)(qrow + kb * 32 + 16 + 8 * hl);
    }
  }

  const v8f vzero = {0.f, 0.f, 0.f, 0.f, 0.f, 0.f, 0.f, 0.f};
  v8f o_acc[8];
  float mrow[8], lrow[8];
#pragma unroll
  for (int i = 0; i < 8; ++i) { o_acc[i] = vzero; mrow[i] = -1e30f; lrow[i] = 0.f; }

  const int blk_end = blockIdx.x * 128 + 127;   // workgroup causal bound
  const int my_end  = q_base + 15;              // this wave's causal bound

  for (int k0 = 0; k0 <= blk_end; k0 += 64) {
    __syncthreads();   // previous tile fully consumed before overwrite
#if HAVE_TDM
    if (wid == 0) {
      // K tile: rows d=0..127 of Kt (inner dim = keys, stride S), 64 keys.
      tdm_load_2d_bf16((unsigned)(uintptr_t)&K_lds[0][0], Kp + k0,
                       S_, D_, 64, 128, (unsigned long long)S_);
      // V tile: rows k0..k0+63 of V (inner dim = d, stride D), full D row.
      tdm_load_2d_bf16((unsigned)(uintptr_t)&V_lds[0][0], Vp + (size_t)k0 * D_,
                       D_, S_, 128, 64, (unsigned long long)D_);
      __builtin_amdgcn_s_wait_tensorcnt(0);
    }
#else
    {
      const int t = threadIdx.x;
#pragma unroll
      for (int i = 0; i < 4; ++i) {
        const int idx = t + i * 256;                 // 0..1023 16B chunks
        const int kr = idx >> 3, kc = (idx & 7) * 8; // K: 8 chunks / row
        *(v4u*)&K_lds[kr][kc] = *(const v4u*)(Kp + (size_t)kr * S_ + k0 + kc);
        const int vr = idx >> 4, vc = (idx & 15) * 8; // V: 16 chunks / row
        *(v4u*)&V_lds[vr][vc] = *(const v4u*)(Vp + (size_t)(k0 + vr) * D_ + vc);
      }
    }
#endif
    __syncthreads();   // tile visible to all waves

    if (k0 > my_end) continue;   // beyond this wave's diagonal: no contribution

    // ---- scores: S = Q (16xD) x K^T (Dx64), 16 WMMAs, B-frags from LDS ----
    v8f sc[4];
#pragma unroll
    for (int nt = 0; nt < 4; ++nt) sc[nt] = vzero;
#pragma unroll
    for (int nt = 0; nt < 4; ++nt) {
#pragma unroll
      for (int kb = 0; kb < 4; ++kb) {
        Frag16 kf;                                   // B frag: lane = key col
        for (int s = 0; s < 8; ++s) {
          kf.e[s]     = K_lds[kb * 32 + 8 * hl + s][nt * 16 + l16];
          kf.e[8 + s] = K_lds[kb * 32 + 16 + 8 * hl + s][nt * 16 + l16];
        }
        sc[nt] = wmma_bf16(qf[kb], kf, sc[nt]);
      }
    }
    // ---- online softmax ---------------------------------------------------
    const bool need_mask = (k0 + 63 > q_base);
    float pv[4][8], rmax[8];
#pragma unroll
    for (int r = 0; r < 8; ++r) {
      const int q_row = q_base + r + 8 * hl;
      float mx = -1e30f;
#pragma unroll
      for (int nt = 0; nt < 4; ++nt) {
        float vv = sc[nt][r];
        if (need_mask && (k0 + nt * 16 + l16) > q_row) vv = -1e30f;  // causal
        pv[nt][r] = vv;
        mx = fmaxf(mx, vv);
      }
      rmax[r] = mx;
    }
#pragma unroll
    for (int off = 8; off >= 1; off >>= 1) {
#pragma unroll
      for (int r = 0; r < 8; ++r)
        rmax[r] = fmaxf(rmax[r], __shfl_xor(rmax[r], off, 32));
    }
    float corr[8];
#pragma unroll
    for (int r = 0; r < 8; ++r) {
      const float mnew = fmaxf(mrow[r], rmax[r]);
      corr[r] = __expf(mrow[r] - mnew);
      mrow[r] = mnew;
      lrow[r] *= corr[r];
    }
#pragma unroll
    for (int t = 0; t < 8; ++t)
#pragma unroll
      for (int r = 0; r < 8; ++r) o_acc[t][r] *= corr[r];

    // exp(P) -> per-wave LDS tile (C layout -> row-major 16x64)
#pragma unroll
    for (int nt = 0; nt < 4; ++nt)
#pragma unroll
      for (int r = 0; r < 8; ++r) {
        const float p = __expf(pv[nt][r] - mrow[r]);
        lrow[r] += p;
        P_lds[wid][r + 8 * hl][nt * 16 + l16] = (__bf16)p;
      }

    // ---- O += P (16x64) x V (64xD), 16 WMMAs ------------------------------
    // DS ops from the same wave are in-order and the P tile is wave-private.
#pragma unroll
    for (int kb2 = 0; kb2 < 2; ++kb2) {
      Frag16 pf;                                     // A frag from LDS
      const __bf16* pr = &P_lds[wid][l16][kb2 * 32 + 8 * hl];
      pf.u4[0] = *(const v4u*)pr;
      pf.u4[1] = *(const v4u*)(pr + 16);
#pragma unroll
      for (int t = 0; t < 8; ++t) {
        Frag16 vf;                                   // B frag: lane = d col
        for (int s = 0; s < 8; ++s) {
          vf.e[s]     = V_lds[kb2 * 32 + 8 * hl + s][t * 16 + l16];
          vf.e[8 + s] = V_lds[kb2 * 32 + 16 + 8 * hl + s][t * 16 + l16];
        }
        o_acc[t] = wmma_bf16(pf, vf, o_acc[t]);
      }
    }
  }

  // ---- finalize: row-sum reduce, normalize, store [B*S, E] bf16 -----------
#pragma unroll
  for (int off = 8; off >= 1; off >>= 1) {
#pragma unroll
    for (int r = 0; r < 8; ++r) lrow[r] += __shfl_xor(lrow[r], off, 32);
  }
  float inv[8];
#pragma unroll
  for (int r = 0; r < 8; ++r) inv[r] = 1.0f / lrow[r];

  const size_t row0 = (size_t)b * S_ + q_base;
#pragma unroll
  for (int t = 0; t < 8; ++t) {
    const int col = h * D_ + t * 16 + l16;
#pragma unroll
    for (int r = 0; r < 8; ++r) {
      const size_t row = row0 + r + 8 * hl;
      O[row * E_ + col] = (__bf16)(o_acc[t][r] * inv[r]);
    }
  }
}

// ----------------------------------------------------------- output proj
__global__ __launch_bounds__(128)
void gemm_out(const __bf16* __restrict__ A, const __bf16* __restrict__ W,
              const float* __restrict__ bias, float* __restrict__ out) {
  const int lane = threadIdx.x & 31;
  const int wid  = threadIdx.x >> 5;
  const int hl   = lane >> 4;
  const int l16  = lane & 15;
  const int m_base = blockIdx.x * 64 + wid * 16;
  const int n_base = blockIdx.y * 64;

  v8f acc[4] = {};
  const __bf16* Arow = A + (size_t)(m_base + l16) * E_;

  for (int k0 = 0; k0 < E_; k0 += 32) {
    Frag16 af;
    af.u4[0] = *(const v4u*)(Arow + k0 + 8 * hl);
    af.u4[1] = *(const v4u*)(Arow + k0 + 16 + 8 * hl);
    const __bf16* Wk0 = W + (size_t)k0 * E_ + n_base + l16;
#pragma unroll
    for (int nt = 0; nt < 4; ++nt) {
      Frag16 bf;
      for (int s = 0; s < 8; ++s) {
        bf.e[s]     = Wk0[(size_t)(8 * hl + s) * E_ + nt * 16];
        bf.e[8 + s] = Wk0[(size_t)(16 + 8 * hl + s) * E_ + nt * 16];
      }
      acc[nt] = wmma_bf16(af, bf, acc[nt]);
    }
  }
#pragma unroll
  for (int nt = 0; nt < 4; ++nt) {
    const int col = n_base + nt * 16 + l16;
    const float bv = bias[col];
#pragma unroll
    for (int r = 0; r < 8; ++r) {
      const int row = m_base + r + 8 * hl;
      out[(size_t)row * E_ + col] = acc[nt][r] + bv;
    }
  }
}

// ---------------------------------------------------------------------------
extern "C" void kernel_launch(void* const* d_in, const int* in_sizes, int n_in,
                              void* d_out, int out_size, void* d_ws, size_t ws_size,
                              hipStream_t stream) {
  (void)in_sizes; (void)n_in; (void)out_size; (void)ws_size;
  const float* query = (const float*)d_in[0];
  const float* key_  = (const float*)d_in[1];
  const float* value = (const float*)d_in[2];
  const float* Wq    = (const float*)d_in[3];
  const float* bq    = (const float*)d_in[4];
  const float* Wk    = (const float*)d_in[5];
  const float* bk    = (const float*)d_in[6];
  const float* Wv    = (const float*)d_in[7];
  const float* bv    = (const float*)d_in[8];
  const float* Wo    = (const float*)d_in[9];
  const float* bo    = (const float*)d_in[10];
  // d_in[11] = causal tril mask; implemented analytically in attn_fwd.
  float* out = (float*)d_out;

  // Workspace layout (bf16): 4 weights (8 MB) + Q/Kt/V (25 MB) + attn (8.4 MB)
  const size_t NW = (size_t)E_ * E_;
  const size_t NX = (size_t)B_ * S_ * E_;
  __bf16* Wq_bf = (__bf16*)d_ws;
  __bf16* Wk_bf = Wq_bf + NW;
  __bf16* Wv_bf = Wk_bf + NW;
  __bf16* Wo_bf = Wv_bf + NW;
  __bf16* Q_bf  = Wo_bf + NW;
  __bf16* Kt_bf = Q_bf + NX;
  __bf16* V_bf  = Kt_bf + NX;
  __bf16* A_bf  = V_bf + NX;

  const int cvtB = (int)((NW + 255) / 256);
  convert_f32_bf16<<<cvtB, 256, 0, stream>>>(Wq, Wq_bf, (int)NW);
  convert_f32_bf16<<<cvtB, 256, 0, stream>>>(Wk, Wk_bf, (int)NW);
  convert_f32_bf16<<<cvtB, 256, 0, stream>>>(Wv, Wv_bf, (int)NW);
  convert_f32_bf16<<<cvtB, 256, 0, stream>>>(Wo, Wo_bf, (int)NW);

  dim3 gGemm(B_ * S_ / 64, E_ / 64, 1);
  const float qscale = 0.08838834764831845f;   // 1/sqrt(128), folded into Q
  gemm_qkv<<<gGemm, 128, 0, stream>>>(query, Wq_bf, bq, Q_bf, 0, qscale);
  gemm_qkv<<<gGemm, 128, 0, stream>>>(key_,  Wk_bf, bk, Kt_bf, 1, 1.0f);
  gemm_qkv<<<gGemm, 128, 0, stream>>>(value, Wv_bf, bv, V_bf, 0, 1.0f);

  attn_fwd<<<dim3(S_ / 128, H_, B_), 256, 0, stream>>>(Q_bf, Kt_bf, V_bf, A_bf);

  gemm_out<<<gGemm, 128, 0, stream>>>(A_bf, Wo_bf, bo, out);
}
